// MultiHeadAttention_412316861010
// MI455X (gfx1250) — matrix operand order repaired and checked by reading, not compiled, not run on hardware
//
#include <hip/hip_runtime.h>

// MHA forward for MI455X (gfx1250): B=2, S=2048, E=1024, H=16, D=64.
// fp32 -> bf16, all GEMMs via v_wmma_f32_16x16x32_bf16 (f32 accumulate),
// flash-attention middle with TDM (tensor_load_to_lds) double-buffered K tiles.

#define E_DIM 1024
#define H_DIM 16
#define D_DIM 64
#define B_DIM 2
#define S_DIM 2048
#define M_ROWS (B_DIM * S_DIM)   // 4096

typedef __attribute__((ext_vector_type(16))) __bf16 bf16x16;
typedef __attribute__((ext_vector_type(8)))  float  f32x8;
typedef __attribute__((ext_vector_type(4)))  unsigned int u32x4;
typedef __attribute__((ext_vector_type(8)))  int i32x8;
typedef __attribute__((ext_vector_type(4)))  int i32x4;

#define HAS_TDM   __has_builtin(__builtin_amdgcn_tensor_load_to_lds)
#define HAS_ASYNC (__has_builtin(__builtin_amdgcn_global_load_async_to_lds_b128) && \
                   __has_builtin(__builtin_amdgcn_s_wait_asynccnt))

union FragBF { bf16x16 v; u32x4 u[2]; };

// round-to-nearest-even fp32 -> bf16 (bit pattern as ushort)
__device__ __forceinline__ unsigned short f2bf(float f) {
  union { float f; unsigned u; } x; x.f = f;
  unsigned r = (x.u + 0x7FFFu + ((x.u >> 16) & 1u)) >> 16;
  return (unsigned short)r;
}

// Load a 16x32 bf16 WMMA operand fragment as two 16-byte runs.
// A-operand: runs at k = 8h and 16+8h (per-lane row = lane%16).
// B-operand: runs at k = 8h and 16+8h (per-lane col = lane%16).
__device__ __forceinline__ bf16x16 ld_frag_2x16(const unsigned short* p0,
                                                const unsigned short* p1) {
  FragBF f;
  f.u[0] = *(const u32x4*)p0;
  f.u[1] = *(const u32x4*)p1;
  return f.v;
}

#if HAS_ASYNC
// 16-byte async copy global -> LDS (ASYNCcnt-tracked).
__device__ __forceinline__ void async_cp16(const void* g, void* l) {
  __builtin_amdgcn_global_load_async_to_lds_b128(
      (__attribute__((address_space(1))) i32x4*)g,
      (__attribute__((address_space(3))) i32x4*)l, 0, 0);
}
#endif

#if HAS_TDM
// Issue a TDM 2-D tile load: 64x64 bf16 tile from a [2048,64] bf16 tensor
// into LDS with padded rows (128B data + 16B pad -> stride 72 elements).
// D# layout per CDNA5 ISA.
__device__ __forceinline__ void tdm_load_k_tile(unsigned lds_off, const void* gaddr) {
  unsigned long long ga = (unsigned long long)(size_t)gaddr;
  u32x4 g0;
  g0.x = 1u;                                   // count=1 (valid), no gather
  g0.y = lds_off;                              // lds_addr (bytes)
  g0.z = (unsigned)(ga & 0xFFFFFFFFu);         // global_addr[31:0]
  g0.w = (unsigned)((ga >> 32) & 0x01FFFFFFu)  // global_addr[56:32]
       | (2u << 30);                           // type = 2 ("image")
  i32x8 g1;
  g1[0] = (int)((1u << 16)       // data_size = 2 bytes
              | (1u << 20)       // pad_enable
              | (4u << 22)       // pad_interval: 32 DWORDs (=128B row)
              | (3u << 25));     // pad_amount: 4 DWORDs (=16B)
  g1[1] = (int)(64u << 16);      // tensor_dim0 = 64  (bits 79:48, low half)
  g1[2] = (int)(2048u << 16);    // tensor_dim1 = 2048 (bits 111:80, low half)
  g1[3] = (int)(64u << 16);      // tile_dim0 = 64 (bits 127:112)
  g1[4] = 64;                    // tile_dim1 = 64 (bits 143:128); tile_dim2 = 0
  g1[5] = 64;                    // tensor_dim0_stride = 64 (bits 191:160)
  g1[6] = 0;                     // dim0_stride hi / dim1_stride lo
  g1[7] = 0;                     // dim1_stride hi
  i32x4 z4 = {0, 0, 0, 0};
  i32x8 z8 = {0, 0, 0, 0, 0, 0, 0, 0};
  // 6-arg form (clang-23 / therock HIP headers): groups 2/3 unused for 2-D.
  __builtin_amdgcn_tensor_load_to_lds(g0, g1, z4, z4, z8, 0);
}
#endif

// Issue the copy of a 64x64 bf16 K tile (row-major, LDS stride 72).
__device__ __forceinline__ void issue_k_tile(unsigned short* dstLds,
                                             const unsigned short* gsrc,
                                             int tid, int wv) {
#if HAS_TDM
  if (wv == 0) tdm_load_k_tile((unsigned)(size_t)dstLds, gsrc);
  (void)tid;
#elif HAS_ASYNC
  int r = tid >> 1, c0 = (tid & 1) * 32;
#pragma unroll
  for (int i = 0; i < 4; ++i)
    async_cp16(gsrc + (size_t)r * D_DIM + c0 + i * 8, dstLds + r * 72 + c0 + i * 8);
  (void)wv;
#else
  int r = tid >> 1, c0 = (tid & 1) * 32;
#pragma unroll
  for (int i = 0; i < 4; ++i)
    *(u32x4*)(dstLds + r * 72 + c0 + i * 8) =
        *(const u32x4*)(gsrc + (size_t)r * D_DIM + c0 + i * 8);
  (void)wv;
#endif
}

// Wait until the current K tile's DMA is complete (double-buffered: when a
// next tile is in flight, one op may remain outstanding).
__device__ __forceinline__ void wait_k_tile(bool hasNext, int wv) {
#if HAS_TDM
  if (wv == 0) {
    if (hasNext) __builtin_amdgcn_s_wait_tensorcnt(1);
    else         __builtin_amdgcn_s_wait_tensorcnt(0);
  }
#elif HAS_ASYNC
  if (hasNext) __builtin_amdgcn_s_wait_asynccnt(4);
  else         __builtin_amdgcn_s_wait_asynccnt(0);
  (void)wv;
#else
  (void)hasNext; (void)wv;
#endif
}

// ---------------------------------------------------------------------------
// GEMM kernel. MODE 0: QKV projection  (A = f32 q/k/v per blockIdx.z,
//                                       W = Wqkv[E,3E], bf16 scatter-store
//                                       to [B,H,S,D]).
//              MODE 1: output project  (A = bf16 ctx [4096,1024],
//                                       W = Wout[E,E], f32 store + bias).
// Block: 256 threads (8 waves). Tile: 128(M) x 64(N), KB = 32.
// ---------------------------------------------------------------------------
template <int MODE>
__global__ __launch_bounds__(256, 2)
void mha_gemm_kernel(const float* __restrict__ Xq, const float* __restrict__ Xk,
                     const float* __restrict__ Xv,
                     const unsigned short* __restrict__ Xbf,
                     const float* __restrict__ Wmat, const float* __restrict__ bias,
                     unsigned short* __restrict__ Dq, unsigned short* __restrict__ Dk,
                     unsigned short* __restrict__ Dv, float* __restrict__ Dout)
{
  constexpr int WN = (MODE == 0) ? 3 * E_DIM : E_DIM;
  __shared__ unsigned short lsA[128 * 40];  // A tile [128][32], stride 40
  __shared__ unsigned short lsB[64 * 40];   // W tile transposed: Bt[n][k]

  const int tid  = threadIdx.x;
  const int lane = tid & 31;
  const int wv   = tid >> 5;
  const int wm   = wv & 3;
  const int wn   = wv >> 2;
  const int hh   = lane >> 4;
  const int lr   = lane & 15;
  const int m0   = blockIdx.y * 128;
  const int z    = blockIdx.z;
  const int ncol0 = blockIdx.x * 64;
  const int gcol0 = ncol0 + ((MODE == 0) ? z * E_DIM : 0);

  const float* X = (MODE == 0) ? ((z == 0) ? Xq : (z == 1) ? Xk : Xv) : nullptr;

  f32x8 acc[2][2] = {};

  for (int kb = 0; kb < E_DIM; kb += 32) {
    // ---- stage A tile into LDS ----
    if (MODE == 0) {
#pragma unroll
      for (int i = 0; i < 4; ++i) {
        int idx = tid + i * 256;
        int r = idx >> 3;
        int c = (idx & 7) << 2;
        float4 f = *(const float4*)(X + (size_t)(m0 + r) * E_DIM + kb + c);
        *(unsigned int*)&lsA[r * 40 + c]     =
            (unsigned)f2bf(f.x) | ((unsigned)f2bf(f.y) << 16);
        *(unsigned int*)&lsA[r * 40 + c + 2] =
            (unsigned)f2bf(f.z) | ((unsigned)f2bf(f.w) << 16);
      }
      if (kb + 32 < E_DIM)
        __builtin_prefetch(X + (size_t)(m0 + (tid >> 1)) * E_DIM + kb + 32, 0, 1);
    } else {
#if HAS_ASYNC
#pragma unroll
      for (int i = 0; i < 2; ++i) {
        int idx = tid + i * 256;
        int r = idx >> 2;
        int c = (idx & 3) << 3;
        async_cp16(Xbf + (size_t)(m0 + r) * E_DIM + kb + c, &lsA[r * 40 + c]);
      }
      __builtin_amdgcn_s_wait_asynccnt(0);
#else
#pragma unroll
      for (int i = 0; i < 2; ++i) {
        int idx = tid + i * 256;
        int r = idx >> 2;
        int c = (idx & 3) << 3;
        *(u32x4*)&lsA[r * 40 + c] =
            *(const u32x4*)(Xbf + (size_t)(m0 + r) * E_DIM + kb + c);
      }
#endif
      if (kb + 32 < E_DIM)
        __builtin_prefetch(Xbf + (size_t)(m0 + (tid >> 1)) * E_DIM + kb + 32, 0, 1);
    }
    // ---- stage W tile transposed into LDS: Bt[n][k] (f32 -> bf16) ----
#pragma unroll
    for (int i = 0; i < 2; ++i) {
      int idx = tid + i * 256;
      int r = idx >> 4;
      int c = (idx & 15) << 2;
      float4 f = *(const float4*)(Wmat + (size_t)(kb + r) * WN + gcol0 + c);
      lsB[(c + 0) * 40 + r] = f2bf(f.x);
      lsB[(c + 1) * 40 + r] = f2bf(f.y);
      lsB[(c + 2) * 40 + r] = f2bf(f.z);
      lsB[(c + 3) * 40 + r] = f2bf(f.w);
    }
    __syncthreads();

    // ---- WMMA compute ----
    bf16x16 a[2], b[2];
#pragma unroll
    for (int mi = 0; mi < 2; ++mi) {
      int row = wm * 32 + mi * 16 + lr;
      a[mi] = ld_frag_2x16(&lsA[row * 40 + 8 * hh],
                           &lsA[row * 40 + 16 + 8 * hh]);
    }
#pragma unroll
    for (int ni = 0; ni < 2; ++ni) {
      int n = wn * 32 + ni * 16 + lr;
      b[ni] = ld_frag_2x16(&lsB[n * 40 + 8 * hh],
                           &lsB[n * 40 + 16 + 8 * hh]);
    }
#pragma unroll
    for (int mi = 0; mi < 2; ++mi)
#pragma unroll
      for (int ni = 0; ni < 2; ++ni)
        acc[mi][ni] = __builtin_amdgcn_wmma_f32_16x16x32_bf16(
            false, a[mi], false, b[ni], (short)0, acc[mi][ni], false, false);
    __syncthreads();
  }

  // ---- epilogue ----
#pragma unroll
  for (int mi = 0; mi < 2; ++mi) {
#pragma unroll
    for (int ni = 0; ni < 2; ++ni) {
#pragma unroll
      for (int r8 = 0; r8 < 8; ++r8) {
        int gm = m0 + wm * 32 + mi * 16 + r8 + 8 * hh;
        int gn = ncol0 + wn * 32 + ni * 16 + lr;
        float vacc = acc[mi][ni][r8];
        if (MODE == 0) {
          vacc += bias[z * E_DIM + gn];
          unsigned short* dst = (z == 0) ? Dq : (z == 1) ? Dk : Dv;
          int bb = gm >> 11;
          int si = gm & (S_DIM - 1);
          int head = gn >> 6;
          int dd = gn & (D_DIM - 1);
          dst[(((size_t)bb * H_DIM + head) * S_DIM + si) * D_DIM + dd] = f2bf(vacc);
        } else {
          Dout[(size_t)gm * E_DIM + gn] = vacc + bias[gn];
        }
      }
    }
  }
}

// ---------------------------------------------------------------------------
// Flash-attention kernel. Block: 128 threads (4 waves); one (b,h) and a
// 64-row Q tile per block; each wave owns 16 Q rows. KV swept in 64-row
// tiles: K tiles streamed by the TDM into double-buffered LDS; V tiles
// transposed into LDS; online softmax; both matmuls via WMMA bf16.
// ---------------------------------------------------------------------------
__global__ __launch_bounds__(128, 2)
void mha_attn_kernel(const unsigned short* __restrict__ Q,
                     const unsigned short* __restrict__ K,
                     const unsigned short* __restrict__ V,
                     const int* __restrict__ mask,
                     unsigned short* __restrict__ Ctx)
{
  __shared__ unsigned short kt[2][64 * 72];   // K tile (row-major), double buffer
  __shared__ unsigned short vt[64 * 72];      // V tile transposed: vt[d][kv]
  __shared__ unsigned short pb[4][16 * 72];   // per-wave P buffer [16 q][64 kv]

  const int tid  = threadIdx.x;
  const int lane = tid & 31;
  const int wv   = tid >> 5;
  const int hh   = lane >> 4;
  const int lr   = lane & 15;
  const int bh   = blockIdx.y;
  const int bb   = bh / H_DIM;
  const int hd   = bh % H_DIM;
  const int q0   = blockIdx.x * 64;

  const unsigned short* Qh = Q + (size_t)bh * S_DIM * D_DIM;
  const unsigned short* Kh = K + (size_t)bh * S_DIM * D_DIM;
  const unsigned short* Vh = V + (size_t)bh * S_DIM * D_DIM;

  // Q fragments for this wave's 16 rows (A-operand layout, from global)
  const int qrow = q0 + wv * 16 + lr;
  bf16x16 aQ[2];
#pragma unroll
  for (int dc = 0; dc < 2; ++dc)
    aQ[dc] = ld_frag_2x16(Qh + (size_t)qrow * D_DIM + dc * 32 + 8 * hh,
                          Qh + (size_t)qrow * D_DIM + dc * 32 + 16 + 8 * hh);

  f32x8 o[4] = {};
  float mrun[8], lrun[8];
#pragma unroll
  for (int r8 = 0; r8 < 8; ++r8) { mrun[r8] = -1e30f; lrun[r8] = 0.f; }

  const float scale = 0.125f;                 // 1/sqrt(D)

  // Preload first K tile into buffer 0.
  issue_k_tile(&kt[0][0], Kh, tid, wv);
  int buf = 0;

  for (int kv0 = 0; kv0 < S_DIM; kv0 += 64) {
    // ---- stage V tile transposed into LDS ----
    {
      int r  = tid >> 1;
      int c0 = (tid & 1) * 32;
#pragma unroll
      for (int i = 0; i < 4; ++i) {
        union { u32x4 u; unsigned short s[8]; } t;
        t.u = *(const u32x4*)(Vh + (size_t)(kv0 + r) * D_DIM + c0 + i * 8);
#pragma unroll
        for (int j = 0; j < 8; ++j)
          vt[(c0 + i * 8 + j) * 72 + r] = t.s[j];
      }
    }
    // ---- pipeline next K tile, wait for current ----
    const bool hasNext = (kv0 + 64) < S_DIM;
    if (hasNext)
      issue_k_tile(&kt[buf ^ 1][0], Kh + (size_t)(kv0 + 64) * D_DIM, tid, wv);
    wait_k_tile(hasNext, wv);
    __syncthreads();
    const unsigned short* kcur = kt[buf];

    // ---- scores = Q . K^T (K rows are the B-operand layout) ----
    f32x8 sacc[4] = {};
#pragma unroll
    for (int dc = 0; dc < 2; ++dc) {
#pragma unroll
      for (int nt = 0; nt < 4; ++nt) {
        bf16x16 bK = ld_frag_2x16(&kcur[(nt * 16 + lr) * 72 + dc * 32 + 8 * hh],
                                  &kcur[(nt * 16 + lr) * 72 + dc * 32 + 16 + 8 * hh]);
        sacc[nt] = __builtin_amdgcn_wmma_f32_16x16x32_bf16(
            false, aQ[dc], false, bK, (short)0, sacc[nt], false, false);
      }
    }

    // ---- scale, mask, online softmax (rows live in 16-lane halves) ----
    const int qg_base = q0 + wv * 16 + 8 * hh;
#pragma unroll
    for (int r8 = 0; r8 < 8; ++r8) {
      int qg = qg_base + r8;
      float mx = -1e30f;
#pragma unroll
      for (int nt = 0; nt < 4; ++nt) {
        int kg = kv0 + nt * 16 + lr;
        float sv = sacc[nt][r8] * scale;
        if (mask[(size_t)qg * S_DIM + kg] == 0) sv = -1e30f;
        sacc[nt][r8] = sv;
        mx = fmaxf(mx, sv);
      }
#pragma unroll
      for (int off = 1; off < 16; off <<= 1)
        mx = fmaxf(mx, __shfl_xor(mx, off, 32));
      float mnew  = fmaxf(mrun[r8], mx);
      float alpha = __expf(mrun[r8] - mnew);
      mrun[r8] = mnew;
      float rs = 0.f;
#pragma unroll
      for (int nt = 0; nt < 4; ++nt) {
        float p = __expf(sacc[nt][r8] - mnew);
        sacc[nt][r8] = p;
        rs += p;
      }
#pragma unroll
      for (int off = 1; off < 16; off <<= 1)
        rs += __shfl_xor(rs, off, 32);
      lrun[r8] = lrun[r8] * alpha + rs;
#pragma unroll
      for (int dt = 0; dt < 4; ++dt) o[dt][r8] *= alpha;
    }

    // ---- re-layout P: C layout -> A-operand layout via per-wave LDS ----
    unsigned short* pw = pb[wv];
#pragma unroll
    for (int r8 = 0; r8 < 8; ++r8)
#pragma unroll
      for (int nt = 0; nt < 4; ++nt)
        pw[(r8 + 8 * hh) * 72 + nt * 16 + lr] = f2bf(sacc[nt][r8]);
    asm volatile("s_wait_dscnt 0" ::: "memory");  // order DS writes before reads

    // ---- O += P . V ----
#pragma unroll
    for (int dc = 0; dc < 2; ++dc) {
      bf16x16 aP = ld_frag_2x16(pw + lr * 72 + dc * 32 + 8 * hh,
                                pw + lr * 72 + dc * 32 + 16 + 8 * hh);
#pragma unroll
      for (int dt = 0; dt < 4; ++dt) {
        bf16x16 bV = ld_frag_2x16(&vt[(dt * 16 + lr) * 72 + dc * 32 + 8 * hh],
                                  &vt[(dt * 16 + lr) * 72 + dc * 32 + 16 + 8 * hh]);
        o[dt] = __builtin_amdgcn_wmma_f32_16x16x32_bf16(
            false, aP, false, bV, (short)0, o[dt], false, false);
      }
    }
    __syncthreads();
    buf ^= 1;
  }

  // ---- normalize and store ctx (bf16, [B*S, E] row-major) ----
#pragma unroll
  for (int r8 = 0; r8 < 8; ++r8) {
    float inv = (lrun[r8] > 0.f) ? (1.f / lrun[r8]) : 0.f;
    int qg = q0 + wv * 16 + r8 + 8 * hh;
#pragma unroll
    for (int dt = 0; dt < 4; ++dt) {
      int dg = hd * D_DIM + dt * 16 + lr;
      Ctx[((size_t)bb * S_DIM + qg) * E_DIM + dg] = f2bf(o[dt][r8] * inv);
    }
  }
}

// ---------------------------------------------------------------------------
extern "C" void kernel_launch(void* const* d_in, const int* in_sizes, int n_in,
                              void* d_out, int out_size, void* d_ws, size_t ws_size,
                              hipStream_t stream) {
  const float* q    = (const float*)d_in[0];
  const float* k    = (const float*)d_in[1];
  const float* v    = (const float*)d_in[2];
  const int*   mask = (const int*)d_in[3];
  const float* Wqkv = (const float*)d_in[4];
  const float* bqkv = (const float*)d_in[5];
  const float* Wout = (const float*)d_in[6];
  const float* bout = (const float*)d_in[7];
  float* out = (float*)d_out;

  const size_t HE = (size_t)B_DIM * H_DIM * S_DIM * D_DIM;  // 4M bf16 elems
  unsigned short* Qbf = (unsigned short*)d_ws;
  unsigned short* Kbf = Qbf + HE;
  unsigned short* Vbf = Kbf + HE;
  unsigned short* Ctx = Vbf + HE;                            // 32 MB total

  dim3 g1(E_DIM / 64, M_ROWS / 128, 3);
  mha_gemm_kernel<0><<<g1, 256, 0, stream>>>(q, k, v, nullptr, Wqkv, bqkv,
                                             Qbf, Kbf, Vbf, nullptr);

  dim3 g2(S_DIM / 64, B_DIM * H_DIM);
  mha_attn_kernel<<<g2, 128, 0, stream>>>(Qbf, Kbf, Vbf, mask, Ctx);

  dim3 g3(E_DIM / 64, M_ROWS / 128, 1);
  mha_gemm_kernel<1><<<g3, 256, 0, stream>>>(nullptr, nullptr, nullptr, Ctx,
                                             Wout, bout, nullptr, nullptr,
                                             nullptr, out);
}
